// Attention_40381282517350
// MI455X (gfx1250) — compile-verified, matrix operand order checked by reading, not run on hardware
//
#include <hip/hip_runtime.h>

// ---------------------------------------------------------------------------
// Types for WMMA fragments (gfx1250, wave32)
// ---------------------------------------------------------------------------
typedef __attribute__((ext_vector_type(16))) __bf16 v16bf;
typedef __attribute__((ext_vector_type(8)))  float  v8f;
typedef int v4i __attribute__((vector_size(16)));   // matches async-LDS builtin

struct __attribute__((aligned(16))) U4 { unsigned int x, y, z, w; };
union Frag { v16bf v; U4 u[2]; };

__device__ inline unsigned short f2bf(float f) {
  union { float f; unsigned int u; } x; x.f = f;
  unsigned int u = x.u;
  u += 0x7fffu + ((u >> 16) & 1u);   // round-to-nearest-even
  return (unsigned short)(u >> 16);
}
__device__ inline float bf2f(unsigned short h) {
  union { unsigned int u; float f; } x; x.u = ((unsigned int)h) << 16;
  return x.f;
}

// ---------------------------------------------------------------------------
// CDNA5 async global->LDS copy (ASYNCcnt path), 16B per call.
// Builtin signature on this toolchain: (v4i AS1* src, AS3 dst, imm off, imm cpol)
// ---------------------------------------------------------------------------
__device__ inline void cp_async_b128(unsigned short* lptr, const unsigned short* gptr) {
#if __has_builtin(__builtin_amdgcn_global_load_async_to_lds_b128)
  __builtin_amdgcn_global_load_async_to_lds_b128(
      (__attribute__((address_space(1))) v4i*)(unsigned long long)gptr,
      (__attribute__((address_space(3))) v4i*)lptr, 0, 0);
#else
  unsigned int loff = (unsigned int)(unsigned long long)
      (__attribute__((address_space(3))) void*)lptr;
  asm volatile("global_load_async_to_lds_b128 %0, %1, off"
               :: "v"(loff), "v"(gptr) : "memory");
#endif
}
__device__ inline void async_wait0() {
#if __has_builtin(__builtin_amdgcn_s_wait_asynccnt)
  __builtin_amdgcn_s_wait_asynccnt(0);
#else
  asm volatile("s_wait_asynccnt 0x0" ::: "memory");
#endif
}

// ---------------------------------------------------------------------------
// Shapes (fixed by the reference)
// ---------------------------------------------------------------------------
#define BATCH 4
#define NTOK  4096
#define CDIM  256
#define BN    (BATCH * NTOK)      // 16384
#define WS    64                  // sqrt(NTOK)
#define ATT_SCALE 0.0625f         // 256^-0.5

// ---------------------------------------------------------------------------
// One 16x16 output tile per wave: D = A(bf16, MxK row-major) * B^T(bf16, NxK)
// ---------------------------------------------------------------------------
__device__ inline v8f wmma_tile(const unsigned short* A, const unsigned short* Bt,
                                int lda, int ldb, int row0, int col0, int K) {
  const int lane = threadIdx.x & 31;
  const int hf   = lane >> 4;
  const int l15  = lane & 15;
  v8f acc = {};
  const unsigned short* ap = A  + (long)(row0 + l15) * lda + 8 * hf;
  const unsigned short* bp = Bt + (long)(col0 + l15) * ldb + 8 * hf;
#pragma unroll
  for (int k = 0; k < K; k += 32) {
    Frag a, b;
    a.u[0] = *(const U4*)(ap + k);
    a.u[1] = *(const U4*)(ap + k + 16);
    b.u[0] = *(const U4*)(bp + k);
    b.u[1] = *(const U4*)(bp + k + 16);
    acc = __builtin_amdgcn_wmma_f32_16x16x32_bf16(false, a.v, false, b.v,
                                                  (short)0, acc, false, false);
  }
  return acc;
}

// ---------------------------------------------------------------------------
// Conversion / transpose helpers
// ---------------------------------------------------------------------------
__global__ void k_cvt_bf16(const float* __restrict__ src,
                           unsigned short* __restrict__ dst, int n) {
  int i = blockIdx.x * 256 + threadIdx.x;
  if (i < n) dst[i] = f2bf(src[i]);
}

// w: [K][N] row-major f32  ->  wt: [N][K] bf16 (K contiguous)
__global__ void k_transpose_bf16(const float* __restrict__ w,
                                 unsigned short* __restrict__ wt, int K, int N) {
  int i = blockIdx.x * 256 + threadIdx.x;
  if (i < K * N) {
    int k = i / N, n = i % N;
    wt[n * K + k] = f2bf(w[i]);
  }
}

// ---------------------------------------------------------------------------
// QKV GEMM: [16384,256] x [256,768].  q,k written [B*N, C]; v written
// transposed [B][C][N] so attention V staging is a contiguous async copy.
// ---------------------------------------------------------------------------
__global__ __launch_bounds__(256) void k_gemm_qkv(
    const unsigned short* __restrict__ xb, const unsigned short* __restrict__ wt,
    unsigned short* __restrict__ qb, unsigned short* __restrict__ kb,
    unsigned short* __restrict__ vT) {
  const int wv   = threadIdx.x >> 5;
  const int row0 = blockIdx.x * 32 + (wv >> 2) * 16;
  const int col0 = blockIdx.y * 64 + (wv & 3) * 16;
  v8f acc = wmma_tile(xb, wt, CDIM, CDIM, row0, col0, CDIM);
  const int lane = threadIdx.x & 31, hf = lane >> 4, l15 = lane & 15;
  const int c = col0 + l15;
  if (c < 512) {                      // uniform across the wave tile
    unsigned short* dst = (c < 256) ? qb : kb;
    const int cc = c & 255;
#pragma unroll
    for (int r = 0; r < 8; ++r) {
      int g = row0 + r + 8 * hf;
      dst[(long)g * CDIM + cc] = f2bf(acc[r]);
    }
  } else {
    const int d = c - 512;
#pragma unroll
    for (int r = 0; r < 8; ++r) {
      int g = row0 + r + 8 * hf;      // g = b*4096 + t
      vT[((long)((g >> 12) * CDIM + d) << 12) + (g & 4095)] = f2bf(acc[r]);
    }
  }
}

// ---------------------------------------------------------------------------
// Gate GEMM + bias + exact GELU -> z (f32)
// ---------------------------------------------------------------------------
__global__ __launch_bounds__(256) void k_gemm_gate(
    const unsigned short* __restrict__ xb, const unsigned short* __restrict__ wt,
    const float* __restrict__ bg, float* __restrict__ z) {
  const int wv   = threadIdx.x >> 5;
  const int row0 = blockIdx.x * 32 + (wv >> 2) * 16;
  const int col0 = blockIdx.y * 64 + (wv & 3) * 16;
  v8f acc = wmma_tile(xb, wt, CDIM, CDIM, row0, col0, CDIM);
  const int lane = threadIdx.x & 31, hf = lane >> 4, l15 = lane & 15;
  const int c = col0 + l15;
  const float bias = bg[c];
#pragma unroll
  for (int r = 0; r < 8; ++r) {
    int g = row0 + r + 8 * hf;
    float v = acc[r] + bias;
    z[(long)g * CDIM + c] = 0.5f * v * (1.0f + erff(v * 0.70710678118654752f));
  }
}

// ---------------------------------------------------------------------------
// Output projection GEMM: y(bf16) x W_proj^T -> out (f32)
// ---------------------------------------------------------------------------
__global__ __launch_bounds__(256) void k_gemm_proj(
    const unsigned short* __restrict__ yb, const unsigned short* __restrict__ wt,
    float* __restrict__ out) {
  const int wv   = threadIdx.x >> 5;
  const int row0 = blockIdx.x * 32 + (wv >> 2) * 16;
  const int col0 = blockIdx.y * 64 + (wv & 3) * 16;
  v8f acc = wmma_tile(yb, wt, CDIM, CDIM, row0, col0, CDIM);
  const int lane = threadIdx.x & 31, hf = lane >> 4, l15 = lane & 15;
  const int c = col0 + l15;
#pragma unroll
  for (int r = 0; r < 8; ++r) {
    int g = row0 + r + 8 * hf;
    out[(long)g * CDIM + c] = acc[r];
  }
}

// ---------------------------------------------------------------------------
// Depthwise 3x3 conv positional encoding on q viewed as [B,C,64,64] -> pe f32
// ---------------------------------------------------------------------------
__global__ void k_pe_conv(const unsigned short* __restrict__ qb,
                          const float* __restrict__ pw, const float* __restrict__ pb,
                          float* __restrict__ pe) {
  int id = blockIdx.x * 256 + threadIdx.x;       // over B*N*C = 4M
  int c = id & 255;
  int n = (id >> 8) & 4095;
  int b = id >> 20;
  int h = n >> 6, w = n & 63;
  float acc = pb[c];
#pragma unroll
  for (int dh = -1; dh <= 1; ++dh) {
#pragma unroll
    for (int dw = -1; dw <= 1; ++dw) {
      int hh = h + dh, ww = w + dw;
      if (hh >= 0 && hh < WS && ww >= 0 && ww < WS) {
        float qv = bf2f(qb[(((long)b << 12) + (hh << 6) + ww) * CDIM + c]);
        acc += qv * pw[c * 9 + (dh + 1) * 3 + (dw + 1)];
      }
    }
  }
  pe[id] = acc;
}

// ---------------------------------------------------------------------------
// Flash attention: block = 128 queries (8 waves x 16 rows), KV tiles of 32.
// Double-buffered async global->LDS staging (ASYNCcnt); K tile key-major,
// V tile pre-transposed in HBM so both copies are contiguous 16B chunks.
// ---------------------------------------------------------------------------
__device__ inline void stage_tiles(const unsigned short* kbuf, const unsigned short* vT,
                                   unsigned short* sK, unsigned short* sVt,
                                   int bb, int b, int kt, int tid) {
  // K tile: 32 keys x 256 d, contiguous 16 KB
  const unsigned short* ks = kbuf + (long)(bb + kt * 32) * CDIM;
#pragma unroll
  for (int it = 0; it < 4; ++it) {
    int off = (tid + it * 256) * 8;
    cp_async_b128(sK + off, ks + off);
  }
  // V^T tile: 256 d-rows x 32 keys (row = 64B = 4 chunks)
#pragma unroll
  for (int it = 0; it < 4; ++it) {
    int i  = tid + it * 256;
    int d  = i >> 2;
    int c8 = (i & 3) * 8;
    cp_async_b128(sVt + d * 32 + c8,
                  vT + ((long)(b * CDIM + d) << 12) + kt * 32 + c8);
  }
}

__global__ __launch_bounds__(256) void k_attn(
    const unsigned short* __restrict__ qb, const unsigned short* __restrict__ kbuf,
    const unsigned short* __restrict__ vT, const float* __restrict__ pe,
    const float* __restrict__ z, unsigned short* __restrict__ y) {
  __shared__ __attribute__((aligned(16))) unsigned short sK[2][32 * CDIM];   // 32 KB
  __shared__ __attribute__((aligned(16))) unsigned short sVt[2][CDIM * 32];  // 32 KB
  __shared__ __attribute__((aligned(16))) unsigned short sP[8][16 * 32];     //  8 KB

  const int tid  = threadIdx.x;
  const int wv   = tid >> 5;
  const int lane = tid & 31;
  const int hf   = lane >> 4;
  const int l15  = lane & 15;
  const int b    = blockIdx.y;
  const int q0   = blockIdx.x * 128 + wv * 16;   // query base within batch
  const int bb   = b << 12;                      // b * 4096

  // Resident Q fragments: 16 rows x 256, 8 chunks of K=32
  Frag qf[8];
  {
    const unsigned short* qp = qb + (long)(bb + q0 + l15) * CDIM + 8 * hf;
#pragma unroll
    for (int ck = 0; ck < 8; ++ck) {
      qf[ck].u[0] = *(const U4*)(qp + ck * 32);
      qf[ck].u[1] = *(const U4*)(qp + ck * 32 + 16);
    }
  }

  v8f zero = {};
  float m[8], lsum[8];
  v8f o[16];
#pragma unroll
  for (int r = 0; r < 8; ++r) { m[r] = -1e30f; lsum[r] = 0.0f; }
#pragma unroll
  for (int f = 0; f < 16; ++f) o[f] = zero;

  stage_tiles(kbuf, vT, sK[0], sVt[0], bb, b, 0, tid);

  for (int kt = 0; kt < NTOK / 32; ++kt) {
    const int cur = kt & 1;
    async_wait0();                               // my tile-kt chunks landed
    __syncthreads();                             // everyone's chunks landed
    if (kt + 1 < NTOK / 32)                      // DMA next tile during compute
      stage_tiles(kbuf, vT, sK[cur ^ 1], sVt[cur ^ 1], bb, b, kt + 1, tid);

    // ---- S = Q K^T (two 16x16 fragments: keys 0-15, 16-31) ----
    v8f s0 = zero, s1 = zero;
    {
      const unsigned short* kp0 = sK[cur] + l15 * CDIM + 8 * hf;
      const unsigned short* kp1 = sK[cur] + (16 + l15) * CDIM + 8 * hf;
#pragma unroll
      for (int ck = 0; ck < 8; ++ck) {
        Frag kb0, kb1;
        kb0.u[0] = *(const U4*)(kp0 + ck * 32);
        kb0.u[1] = *(const U4*)(kp0 + ck * 32 + 16);
        kb1.u[0] = *(const U4*)(kp1 + ck * 32);
        kb1.u[1] = *(const U4*)(kp1 + ck * 32 + 16);
        s0 = __builtin_amdgcn_wmma_f32_16x16x32_bf16(false, qf[ck].v, false, kb0.v,
                                                     (short)0, s0, false, false);
        s1 = __builtin_amdgcn_wmma_f32_16x16x32_bf16(false, qf[ck].v, false, kb1.v,
                                                     (short)0, s1, false, false);
      }
    }

    // ---- online softmax (row = r + 8*hf; 16 cols live in this lane-half) ----
    unsigned short* pw = &sP[wv][0];
#pragma unroll
    for (int r = 0; r < 8; ++r) {
      float a0 = s0[r] * ATT_SCALE, a1 = s1[r] * ATT_SCALE;
      float mx = fmaxf(a0, a1);
#pragma unroll
      for (int off = 8; off >= 1; off >>= 1) mx = fmaxf(mx, __shfl_xor(mx, off, 32));
      float mn = fmaxf(m[r], mx);
      float alpha = __expf(m[r] - mn);
      float p0 = __expf(a0 - mn), p1 = __expf(a1 - mn);
      float rs = p0 + p1;
#pragma unroll
      for (int off = 8; off >= 1; off >>= 1) rs += __shfl_xor(rs, off, 32);
      lsum[r] = lsum[r] * alpha + rs;
      m[r] = mn;
#pragma unroll
      for (int f = 0; f < 16; ++f) o[f][r] *= alpha;
      int rr = r + 8 * hf;
      pw[rr * 32 + l15]      = f2bf(p0);
      pw[rr * 32 + 16 + l15] = f2bf(p1);
    }

    asm volatile("s_wait_dscnt 0x0" ::: "memory");  // wave-local P round-trip

    // ---- re-fragment P (16x32) as WMMA A operand ----
    Frag pf;
    {
      const unsigned short* pp = pw + l15 * 32 + 8 * hf;
      pf.u[0] = *(const U4*)(pp);
      pf.u[1] = *(const U4*)(pp + 16);
    }
    // ---- O += P * V over 16 d-column fragments ----
#pragma unroll
    for (int f = 0; f < 16; ++f) {
      Frag vf;
      const unsigned short* vp = sVt[cur] + (f * 16 + l15) * 32 + 8 * hf;
      vf.u[0] = *(const U4*)(vp);
      vf.u[1] = *(const U4*)(vp + 16);
      o[f] = __builtin_amdgcn_wmma_f32_16x16x32_bf16(false, pf.v, false, vf.v,
                                                     (short)0, o[f], false, false);
    }
    __syncthreads();
  }

  // ---- epilogue: O/l + pe, * z, store bf16 y ----
#pragma unroll
  for (int r = 0; r < 8; ++r) {
    float inv = 1.0f / lsum[r];
    int g = bb + q0 + r + 8 * hf;
#pragma unroll
    for (int f = 0; f < 16; ++f) {
      int c = f * 16 + l15;
      long idx = (long)g * CDIM + c;
      float val = o[f][r] * inv + pe[idx];
      val *= z[idx];
      y[idx] = f2bf(val);
    }
  }
}

// ---------------------------------------------------------------------------
// Host-side launch
// ---------------------------------------------------------------------------
extern "C" void kernel_launch(void* const* d_in, const int* in_sizes, int n_in,
                              void* d_out, int out_size, void* d_ws, size_t ws_size,
                              hipStream_t stream) {
  (void)in_sizes; (void)n_in; (void)out_size; (void)ws_size;
  const float* x      = (const float*)d_in[0];
  const float* W_qkv  = (const float*)d_in[1];
  const float* W_gate = (const float*)d_in[2];
  const float* b_gate = (const float*)d_in[3];
  const float* W_proj = (const float*)d_in[4];
  const float* pe_w   = (const float*)d_in[5];
  const float* pe_b   = (const float*)d_in[6];
  float* out = (float*)d_out;

  char* ws = (char*)d_ws;
  size_t off = 0;
  auto alloc = [&](size_t bytes) -> void* {
    void* p = ws + off;
    off += (bytes + 255) & ~(size_t)255;
    return p;
  };
  unsigned short* xb     = (unsigned short*)alloc((size_t)BN * CDIM * 2);
  unsigned short* wqkvt  = (unsigned short*)alloc((size_t)768 * CDIM * 2);
  unsigned short* wgatet = (unsigned short*)alloc((size_t)CDIM * CDIM * 2);
  unsigned short* wprojt = (unsigned short*)alloc((size_t)CDIM * CDIM * 2);
  unsigned short* qB     = (unsigned short*)alloc((size_t)BN * CDIM * 2);
  unsigned short* kB     = (unsigned short*)alloc((size_t)BN * CDIM * 2);
  unsigned short* vTB    = (unsigned short*)alloc((size_t)BN * CDIM * 2);
  float*          zB     = (float*)alloc((size_t)BN * CDIM * 4);
  float*          peB    = (float*)alloc((size_t)BN * CDIM * 4);
  unsigned short* yB     = (unsigned short*)alloc((size_t)BN * CDIM * 2);

  // 1) conversions / weight transposes
  k_cvt_bf16<<<(BN * CDIM) / 256, 256, 0, stream>>>(x, xb, BN * CDIM);
  k_transpose_bf16<<<(CDIM * 768) / 256, 256, 0, stream>>>(W_qkv, wqkvt, CDIM, 768);
  k_transpose_bf16<<<(CDIM * CDIM) / 256, 256, 0, stream>>>(W_gate, wgatet, CDIM, CDIM);
  k_transpose_bf16<<<(CDIM * CDIM) / 256, 256, 0, stream>>>(W_proj, wprojt, CDIM, CDIM);
  // 2) QKV GEMM (v stored transposed [B][C][N])
  k_gemm_qkv<<<dim3(BN / 32, 768 / 64), 256, 0, stream>>>(xb, wqkvt, qB, kB, vTB);
  // 3) gate GEMM + GELU
  k_gemm_gate<<<dim3(BN / 32, CDIM / 64), 256, 0, stream>>>(xb, wgatet, b_gate, zB);
  // 4) depthwise-conv positional encoding (needs q)
  k_pe_conv<<<(BN * CDIM) / 256, 256, 0, stream>>>(qB, pe_w, pe_b, peB);
  // 5) flash attention, async double-buffered KV staging, fused (+pe)*z
  k_attn<<<dim3(NTOK / 128, BATCH), 256, 0, stream>>>(qB, kB, vTB, peB, zB, yB);
  // 6) output projection
  k_gemm_proj<<<dim3(BN / 32, CDIM / 64), 256, 0, stream>>>(yB, wprojt, out);
}
